// DynSMHALayer_16853451670036
// MI455X (gfx1250) — matrix operand order repaired and checked by reading, not compiled, not run on hardware
//
#include <hip/hip_runtime.h>
#include <hip/hip_bf16.h>
#include <math.h>

typedef __bf16 bf16x16 __attribute__((ext_vector_type(16)));
typedef float  f32x8   __attribute__((ext_vector_type(8)));
typedef unsigned short u16;

union Frag {
  bf16x16 v;
  unsigned u[8];
};

// ---- optional gfx1250 async global->LDS path (guarded; falls back cleanly) ----
#if defined(__has_builtin)
#if __has_builtin(__builtin_amdgcn_global_load_async_to_lds_b64) && \
    __has_builtin(__builtin_amdgcn_s_wait_asynccnt)
#define USE_ASYNC_LDS 1
#endif
#endif
#ifndef USE_ASYNC_LDS
#define USE_ASYNC_LDS 0
#endif

#if USE_ASYNC_LDS
typedef int v2i __attribute__((vector_size(8)));
// builtin signature: (v2i AS1* src, v2i AS3* dst, imm int offset, imm int cpol)
#define GLB_PTR(p) ((__attribute__((address_space(1))) v2i*)(p))
#define LDS_PTR(p) ((__attribute__((address_space(3))) v2i*)(p))
#endif

__device__ __forceinline__ u16 f2bf(float f) {
  unsigned u = __float_as_uint(f);
  return (u16)((u + 0x7fffu + ((u >> 16) & 1u)) >> 16);  // RNE
}

__device__ __forceinline__ f32x8 wmma_bf16(const Frag& a, const Frag& b, f32x8 c) {
  // D = A(16x32 bf16) * B(32x16 bf16) + C(16x16 f32)
  return __builtin_amdgcn_wmma_f32_16x16x32_bf16(false, a.v, false, b.v,
                                                 (short)0, c, false, false);
}

// ---------------------------------------------------------------- convert f32 -> bf16
__global__ void cvt_kernel(const float* __restrict__ in, u16* __restrict__ out, int n) {
  for (int i = blockIdx.x * 256 + threadIdx.x; i < n; i += gridDim.x * 256)
    out[i] = f2bf(in[i]);
}

// ---------------------------------------------------------------- gating prep
// prep[0..15]  = 1/sqrt(colnorm^2(sim)+1e-12), prep[16..31] = sigmoid(gates)
__global__ void prep_kernel(const float* __restrict__ sim, const float* __restrict__ gates,
                            float* __restrict__ prep) {
  int e = threadIdx.x;
  if (e < 16) {
    float s = 0.f;
    for (int c = 0; c < 1024; ++c) { float v = sim[c * 16 + e]; s += v * v; }
    prep[e]      = 1.f / sqrtf(s + 1e-12f);
    prep[16 + e] = 1.f / (1.f + expf(-gates[e]));
  }
}

// ---------------------------------------------------------------- gating: one wave per token
__global__ void gating_kernel(const float* __restrict__ hidden, const float* __restrict__ sim,
                              const float* __restrict__ prep, float* __restrict__ rw) {
  const int wave = threadIdx.x >> 5, lane = threadIdx.x & 31;
  const int token = blockIdx.x * 8 + wave;
  const float* h = hidden + (size_t)token * 1024;

  float dots[16];
#pragma unroll
  for (int e = 0; e < 16; ++e) dots[e] = 0.f;
  float hh = 0.f;

  for (int c = lane; c < 1024; c += 32) {
    float hv = h[c];
    hh += hv * hv;
    const float* sr = sim + (size_t)c * 16;
#pragma unroll
    for (int e = 0; e < 16; ++e) dots[e] += hv * sr[e];
  }
#pragma unroll
  for (int m = 16; m >= 1; m >>= 1) {
    hh += __shfl_xor(hh, m, 32);
#pragma unroll
    for (int e = 0; e < 16; ++e) dots[e] += __shfl_xor(dots[e], m, 32);
  }

  if (lane == 0) {
    float rh = 1.f / sqrtf(hh + 1e-12f);
    float logit[16], gated[16];
    bool mask[16];
    int nact = 0;
    for (int e = 0; e < 16; ++e) {
      logit[e] = dots[e] * rh * prep[e] - prep[16 + e];
      gated[e] = logit[e] > 0.f ? logit[e] : 0.f;
      mask[e]  = logit[e] > 0.f;
      nact += mask[e] ? 1 : 0;
    }
    if (nact == 0) {  // top-2 fallback on raw logits
      int i1 = 0; float b1 = -1e30f;
      for (int e = 0; e < 16; ++e) if (logit[e] > b1) { b1 = logit[e]; i1 = e; }
      int i2 = 0; float b2 = -1e31f;
      for (int e = 0; e < 16; ++e) if (e != i1 && logit[e] > b2) { b2 = logit[e]; i2 = e; }
      mask[i1] = true; mask[i2] = true;
    }
    float mx = -1e30f;
    for (int e = 0; e < 16; ++e) if (mask[e] && gated[e] > mx) mx = gated[e];
    float p[16], s = 0.f;
    for (int e = 0; e < 16; ++e) { p[e] = mask[e] ? expf(gated[e] - mx) : 0.f; s += p[e]; }
    float inv = 1.f / s;
    for (int e = 0; e < 16; ++e) rw[(size_t)token * 16 + e] = p[e] * inv;
  }
}

// ---------------------------------------------------------------- fused expert GEMM
// C[m,n] = sum_e rw[m,e] * (A[m,:] @ W_e[:,n]);  A bf16 [M x K] (lda=K),
// W bf16 per-expert [K x ldw] (expert stride wStride). Block tile 128x64,
// 8 waves in 4x2, wave tile 32x32 (2x2 WMMA 16x16x32 tiles).
__global__ __launch_bounds__(256)
void moe_gemm_kernel(const u16* __restrict__ A, int lda,
                     const u16* __restrict__ W, int ldw, int wStride,
                     const float* __restrict__ rw,
                     void* __restrict__ Cout, int ldc, int outBf16, int K) {
  __shared__ u16 lds_a[128 * 36];  // [m][k], k-stride 36 (72B, 8B aligned)
  __shared__ u16 lds_b[64 * 36];   // [n][k]
  const int tid = threadIdx.x;
  const int wave = tid >> 5, lane = tid & 31;
  const int l15 = lane & 15, lh = lane >> 4;
  const int wm = wave >> 1, wn = wave & 1;
  const int m0 = blockIdx.y * 128, n0 = blockIdx.x * 64;
  const int kbA = lh * 16;  // A frag: lanes 16-31 start at K=16
  const int kbB = lh * 16;  // B frag: lanes 16-31 start at K=16

  const f32x8 fz = {0.f, 0.f, 0.f, 0.f, 0.f, 0.f, 0.f, 0.f};
  f32x8 tot[2][2] = {{fz, fz}, {fz, fz}};

  for (int e = 0; e < 16; ++e) {
    f32x8 acc[2][2] = {{fz, fz}, {fz, fz}};
    const u16* We = W + (size_t)e * wStride;
    for (int k0 = 0; k0 < K; k0 += 32) {
      __syncthreads();
      // ---- A tile 128x32: 1024 chunks of 4 bf16 (4 per thread) ----
#if USE_ASYNC_LDS
#pragma unroll
      for (int it = 0; it < 4; ++it) {
        int c = tid + it * 256;
        int row = c >> 3, kk = (c & 7) << 2;
        __builtin_amdgcn_global_load_async_to_lds_b64(
            GLB_PTR(A + (size_t)(m0 + row) * lda + k0 + kk),
            LDS_PTR(&lds_a[row * 36 + kk]), 0, 0);
      }
#else
      {
        uint2 va[4];
#pragma unroll
        for (int it = 0; it < 4; ++it) {
          int c = tid + it * 256;
          int row = c >> 3, kk = (c & 7) << 2;
          va[it] = *(const uint2*)(A + (size_t)(m0 + row) * lda + k0 + kk);
        }
#pragma unroll
        for (int it = 0; it < 4; ++it) {
          int c = tid + it * 256;
          int row = c >> 3, kk = (c & 7) << 2;
          *(uint2*)(&lds_a[row * 36 + kk]) = va[it];
        }
      }
#endif
      // ---- B tile 32x64, transposed into [n][k] (2 chunks per thread) ----
      {
        uint2 vb[2];
#pragma unroll
        for (int it = 0; it < 2; ++it) {
          int c = tid + it * 256;
          int kr = c >> 4, nn = (c & 15) << 2;
          vb[it] = *(const uint2*)(We + (size_t)(k0 + kr) * ldw + n0 + nn);
        }
#pragma unroll
        for (int it = 0; it < 2; ++it) {
          int c = tid + it * 256;
          int kr = c >> 4, nn = (c & 15) << 2;
          lds_b[(nn + 0) * 36 + kr] = (u16)(vb[it].x & 0xffffu);
          lds_b[(nn + 1) * 36 + kr] = (u16)(vb[it].x >> 16);
          lds_b[(nn + 2) * 36 + kr] = (u16)(vb[it].y & 0xffffu);
          lds_b[(nn + 3) * 36 + kr] = (u16)(vb[it].y >> 16);
        }
      }
      // prefetch next k-step tiles into cache
      if (k0 + 32 < K) {
        __builtin_prefetch(A + (size_t)(m0 + (tid >> 1)) * lda + k0 + 32 + (tid & 1) * 16, 0, 0);
        __builtin_prefetch(We + (size_t)(k0 + 32 + (tid >> 3)) * ldw + n0 + (tid & 7) * 8, 0, 0);
      }
#if USE_ASYNC_LDS
      __builtin_amdgcn_s_wait_asynccnt(0);
#endif
      __syncthreads();
      Frag af[2], bf[2];
#pragma unroll
      for (int mi = 0; mi < 2; ++mi) {
        int row = wm * 32 + mi * 16 + l15;
#pragma unroll
        for (int p = 0; p < 8; ++p) {
          int kk = kbA + 2 * p;
          af[mi].u[p] = *(const unsigned*)(&lds_a[row * 36 + kk]);
        }
      }
#pragma unroll
      for (int ni = 0; ni < 2; ++ni) {
        int col = wn * 32 + ni * 16 + l15;
#pragma unroll
        for (int p = 0; p < 8; ++p)
          bf[ni].u[p] = *(const unsigned*)(&lds_b[col * 36 + kbB + 2 * p]);
      }
#pragma unroll
      for (int mi = 0; mi < 2; ++mi)
#pragma unroll
        for (int ni = 0; ni < 2; ++ni)
          acc[mi][ni] = wmma_bf16(af[mi], bf[ni], acc[mi][ni]);
    }
    // expert epilogue: total += rw[row,e] * acc
#pragma unroll
    for (int mi = 0; mi < 2; ++mi)
#pragma unroll
      for (int r = 0; r < 8; ++r) {
        int grow = m0 + wm * 32 + mi * 16 + r + 8 * lh;
        float wgt = rw[(size_t)grow * 16 + e];
        tot[mi][0][r] += wgt * acc[mi][0][r];
        tot[mi][1][r] += wgt * acc[mi][1][r];
      }
  }
  // store
  if (outBf16) {
    u16* Cb = (u16*)Cout;
#pragma unroll
    for (int mi = 0; mi < 2; ++mi)
#pragma unroll
      for (int ni = 0; ni < 2; ++ni)
#pragma unroll
        for (int r = 0; r < 8; ++r) {
          int grow = m0 + wm * 32 + mi * 16 + r + 8 * lh;
          int col = n0 + wn * 32 + ni * 16 + l15;
          Cb[(size_t)grow * ldc + col] = f2bf(tot[mi][ni][r]);
        }
  } else {
    float* Cf = (float*)Cout;
#pragma unroll
    for (int mi = 0; mi < 2; ++mi)
#pragma unroll
      for (int ni = 0; ni < 2; ++ni)
#pragma unroll
        for (int r = 0; r < 8; ++r) {
          int grow = m0 + wm * 32 + mi * 16 + r + 8 * lh;
          int col = n0 + wn * 32 + ni * 16 + l15;
          Cf[(size_t)grow * ldc + col] = tot[mi][ni][r];
        }
  }
}

// ---------------------------------------------------------------- causal flash attention
// 8 waves x 16 q-rows per block; shared K/V tiles of 32 keys; online softmax.
__global__ void fa_kernel(const u16* __restrict__ Q, const u16* __restrict__ K,
                          const u16* __restrict__ V, u16* __restrict__ O) {
  const int T = 2048, HD = 128;
  const int b = blockIdx.y, q0 = blockIdx.x * 128;
  const int tid = threadIdx.x, wave = tid >> 5, lane = tid & 31;
  const int l15 = lane & 15, lh = lane >> 4;
  const int qr = q0 + wave * 16;
  const int kbA = lh * 16, kbB = lh * 16;

  __shared__ u16 lds_k[32 * 136];     // [key][ch]
  __shared__ u16 lds_v[128 * 36];     // [ch][key] (transposed)
  __shared__ u16 lds_p[8][16 * 36];   // per-wave P bounce [row][key]

  const u16* Qb = Q + (size_t)b * T * HD;
  const u16* Kb = K + (size_t)b * T * HD;
  const u16* Vb = V + (size_t)b * T * HD;

  // resident Q fragments (16 rows x 128 ch -> 4 k-steps)
  Frag qf[4];
  {
    int qrow = qr + l15;
#pragma unroll
    for (int s = 0; s < 4; ++s)
#pragma unroll
      for (int p = 0; p < 8; ++p) {
        int kk = kbA + 2 * p;
        qf[s].u[p] = *(const unsigned*)(Qb + (size_t)qrow * HD + s * 32 + kk);
      }
  }
  const f32x8 fz = {0.f, 0.f, 0.f, 0.f, 0.f, 0.f, 0.f, 0.f};
  f32x8 oacc[8] = {fz, fz, fz, fz, fz, fz, fz, fz};
  float mS[8], lS[8];
#pragma unroll
  for (int r = 0; r < 8; ++r) { mS[r] = -1e30f; lS[r] = 0.f; }

  const float scale = 0.08838834764831845f;  // 1/sqrt(128)
  const int nsteps = (q0 + 128) / 32;

  for (int st = 0; st < nsteps; ++st) {
    const int k0 = st * 32;
    __syncthreads();
    // ---- K tile 32x128 (4 chunks per thread) ----
#if USE_ASYNC_LDS
#pragma unroll
    for (int it = 0; it < 4; ++it) {
      int c = tid + it * 256;
      int key = c >> 5, ch = (c & 31) << 2;
      __builtin_amdgcn_global_load_async_to_lds_b64(
          GLB_PTR(Kb + (size_t)(k0 + key) * HD + ch),
          LDS_PTR(&lds_k[key * 136 + ch]), 0, 0);
    }
#else
    {
      uint2 vk[4];
#pragma unroll
      for (int it = 0; it < 4; ++it) {
        int c = tid + it * 256;
        int key = c >> 5, ch = (c & 31) << 2;
        vk[it] = *(const uint2*)(Kb + (size_t)(k0 + key) * HD + ch);
      }
#pragma unroll
      for (int it = 0; it < 4; ++it) {
        int c = tid + it * 256;
        int key = c >> 5, ch = (c & 31) << 2;
        *(uint2*)(&lds_k[key * 136 + ch]) = vk[it];
      }
    }
#endif
    // ---- V tile 32x128, transposed (4 chunks per thread) ----
    {
      uint2 vv[4];
#pragma unroll
      for (int it = 0; it < 4; ++it) {
        int c = tid + it * 256;
        int key = c >> 5, ch = (c & 31) << 2;
        vv[it] = *(const uint2*)(Vb + (size_t)(k0 + key) * HD + ch);
      }
#pragma unroll
      for (int it = 0; it < 4; ++it) {
        int c = tid + it * 256;
        int key = c >> 5, ch = (c & 31) << 2;
        lds_v[(ch + 0) * 36 + key] = (u16)(vv[it].x & 0xffffu);
        lds_v[(ch + 1) * 36 + key] = (u16)(vv[it].x >> 16);
        lds_v[(ch + 2) * 36 + key] = (u16)(vv[it].y & 0xffffu);
        lds_v[(ch + 3) * 36 + key] = (u16)(vv[it].y >> 16);
      }
    }
    if (st + 1 < nsteps) {
      __builtin_prefetch(Kb + (size_t)(k0 + 32 + (tid >> 3)) * HD + (tid & 7) * 16, 0, 0);
      __builtin_prefetch(Vb + (size_t)(k0 + 32 + (tid >> 3)) * HD + (tid & 7) * 16, 0, 0);
    }
#if USE_ASYNC_LDS
    __builtin_amdgcn_s_wait_asynccnt(0);
#endif
    __syncthreads();
    if (k0 > qr + 15) continue;  // fully-masked step for this wave (after barriers)

    // S = Q K^T  (two 16-key n-tiles, K-dim = 128 channels)
    f32x8 sacc[2] = {fz, fz};
#pragma unroll
    for (int nt = 0; nt < 2; ++nt) {
      int key = nt * 16 + l15;
#pragma unroll
      for (int s = 0; s < 4; ++s) {
        Frag kf;
#pragma unroll
        for (int p = 0; p < 8; ++p)
          kf.u[p] = *(const unsigned*)(&lds_k[key * 136 + s * 32 + kbB + 2 * p]);
        sacc[nt] = wmma_bf16(qf[s], kf, sacc[nt]);
      }
    }
    // scale + causal mask + online softmax stats
    float alpha[8];
#pragma unroll
    for (int r = 0; r < 8; ++r) {
      int row = qr + r + 8 * lh;
#pragma unroll
      for (int nt = 0; nt < 2; ++nt) {
        int key = k0 + nt * 16 + l15;
        float v = sacc[nt][r] * scale;
        if (key > row) v = -1e30f;
        sacc[nt][r] = v;
      }
      float rm = fmaxf(sacc[0][r], sacc[1][r]);
#pragma unroll
      for (int m = 8; m >= 1; m >>= 1) rm = fmaxf(rm, __shfl_xor(rm, m, 32));
      float mn = fmaxf(mS[r], rm);
      alpha[r] = expf(mS[r] - mn);
      mS[r] = mn;
      float p0 = expf(sacc[0][r] - mn), p1 = expf(sacc[1][r] - mn);
      sacc[0][r] = p0; sacc[1][r] = p1;
      float rs = p0 + p1;
#pragma unroll
      for (int m = 8; m >= 1; m >>= 1) rs += __shfl_xor(rs, m, 32);
      lS[r] = lS[r] * alpha[r] + rs;
    }
#pragma unroll
    for (int ni = 0; ni < 8; ++ni)
#pragma unroll
      for (int r = 0; r < 8; ++r) oacc[ni][r] *= alpha[r];

    // P: C-layout -> LDS -> A-layout
#pragma unroll
    for (int nt = 0; nt < 2; ++nt)
#pragma unroll
      for (int r = 0; r < 8; ++r)
        lds_p[wave][(r + 8 * lh) * 36 + nt * 16 + l15] = f2bf(sacc[nt][r]);
    Frag pf;
#pragma unroll
    for (int p = 0; p < 8; ++p) {
      int kk = kbA + 2 * p;
      pf.u[p] = *(const unsigned*)(&lds_p[wave][l15 * 36 + kk]);
    }
    // O += P V   (8 channel tiles, K-dim = 32 keys)
#pragma unroll
    for (int ni = 0; ni < 8; ++ni) {
      Frag vf;
      int ch = ni * 16 + l15;
#pragma unroll
      for (int p = 0; p < 8; ++p)
        vf.u[p] = *(const unsigned*)(&lds_v[ch * 36 + kbB + 2 * p]);
      oacc[ni] = wmma_bf16(pf, vf, oacc[ni]);
    }
  }
  // normalize + write bf16
#pragma unroll
  for (int ni = 0; ni < 8; ++ni)
#pragma unroll
    for (int r = 0; r < 8; ++r) {
      int row = qr + r + 8 * lh;
      O[((size_t)b * T + row) * HD + ni * 16 + l15] = f2bf(oacc[ni][r] / lS[r]);
    }
}

// ---------------------------------------------------------------- launch
extern "C" void kernel_launch(void* const* d_in, const int* in_sizes, int n_in,
                              void* d_out, int out_size, void* d_ws, size_t ws_size,
                              hipStream_t stream) {
  (void)in_sizes; (void)n_in; (void)out_size; (void)ws_size;
  const float* hidden = (const float*)d_in[0];
  const float* sim    = (const float*)d_in[1];
  const float* gates  = (const float*)d_in[2];
  const float* qp     = (const float*)d_in[3];
  const float* kp     = (const float*)d_in[4];
  const float* vp     = (const float*)d_in[5];
  const float* op     = (const float*)d_in[6];
  float* out = (float*)d_out;

  const int BT = 8192, HID = 1024, HD = 128, E = 16;

  char* ws = (char*)d_ws;
  size_t off = 0;
  auto alloc = [&](size_t bytes) -> void* {
    void* p = ws + off;
    off += (bytes + 255) & ~(size_t)255;
    return p;
  };
  u16*   hB   = (u16*)alloc((size_t)BT * HID * 2);
  u16*   WqB  = (u16*)alloc((size_t)E * HID * HD * 2);
  u16*   WkB  = (u16*)alloc((size_t)E * HID * HD * 2);
  u16*   WvB  = (u16*)alloc((size_t)E * HID * HD * 2);
  u16*   WoB  = (u16*)alloc((size_t)E * HD * HID * 2);
  float* prep = (float*)alloc(32 * sizeof(float));
  float* rw   = (float*)alloc((size_t)BT * E * sizeof(float));
  u16*   qB   = (u16*)alloc((size_t)BT * HD * 2);
  u16*   kB   = (u16*)alloc((size_t)BT * HD * 2);
  u16*   vB   = (u16*)alloc((size_t)BT * HD * 2);
  u16*   aoB  = (u16*)alloc((size_t)BT * HD * 2);

  cvt_kernel<<<4096, 256, 0, stream>>>(hidden, hB, BT * HID);
  cvt_kernel<<<2048, 256, 0, stream>>>(qp, WqB, E * HID * HD);
  cvt_kernel<<<2048, 256, 0, stream>>>(kp, WkB, E * HID * HD);
  cvt_kernel<<<2048, 256, 0, stream>>>(vp, WvB, E * HID * HD);
  cvt_kernel<<<2048, 256, 0, stream>>>(op, WoB, E * HD * HID);

  prep_kernel<<<1, 32, 0, stream>>>(sim, gates, prep);
  gating_kernel<<<BT / 8, 256, 0, stream>>>(hidden, sim, prep, rw);

  // Q/K/V: A=[8192x1024] bf16, per-expert W=[1024x128], fused rw-combine, bf16 out
  moe_gemm_kernel<<<dim3(HD / 64, BT / 128), 256, 0, stream>>>(
      hB, HID, WqB, HD, HID * HD, rw, qB, HD, 1, HID);
  moe_gemm_kernel<<<dim3(HD / 64, BT / 128), 256, 0, stream>>>(
      hB, HID, WkB, HD, HID * HD, rw, kB, HD, 1, HID);
  moe_gemm_kernel<<<dim3(HD / 64, BT / 128), 256, 0, stream>>>(
      hB, HID, WvB, HD, HID * HD, rw, vB, HD, 1, HID);

  // causal flash attention: grid (T/128, B)
  fa_kernel<<<dim3(16, 4), 256, 0, stream>>>(qB, kB, vB, aoB);

  // output projection: A=ao [8192x128] bf16, per-expert W=[128x1024], f32 out
  moe_gemm_kernel<<<dim3(HID / 64, BT / 128), 256, 0, stream>>>(
      aoB, HD, WoB, HID, HD * HID, rw, out, HID, 0, HD);
}
